// FrustumPointNet_49744311222690
// MI455X (gfx1250) — matrix operand order repaired and checked by reading, not compiled, not run on hardware
//
#include <hip/hip_runtime.h>
#include <hip/hip_bf16.h>
#include <hip/hip_fp16.h>
#include <stdint.h>

typedef __attribute__((ext_vector_type(16))) _Float16 v16h;
typedef __attribute__((ext_vector_type(8)))  _Float16 v8h;
typedef __attribute__((ext_vector_type(8)))  float    v8f;

#define B_SZ  128
#define N_PTS 1024
#define NSEG  512

#define TM 128   // M rows per block
#define TN 64    // N cols (points) per block
#define TK 32    // K per step
#define LDK 40   // padded LDS row stride in halfs (80B = 5x16B, keeps b128 alignment)

// epilogue flags
#define F_F16  1
#define F_F32  2
#define F_POOL 4
#define F_RELU 8

union FragH { v16h v; v8h h[2]; };

// CDNA5 async global->LDS staging (ASYNCcnt-tracked, overlaps WMMA compute)
__device__ inline void async_ld_b128(unsigned lds, unsigned long long gaddr) {
  asm volatile("global_load_async_to_lds_b128 %0, %1, off"
               :: "v"(lds), "v"(gaddr) : "memory");
}
__device__ inline void wait_async0() {
  asm volatile("s_wait_asynccnt 0x0" ::: "memory");
}

// ---------------------------------------------------------------------------
// f16 WMMA GEMM, point-major activations:
//   out[p][m] = sum_k W[m][k] * X[p][k]  (+bias[m] +extra[batch][m])
// W: [Opad][Cpad] (O zero-padded to 128, C to 32). X: [P][Cpad] per batch.
// grid = (Opad/128, P/64, nBatch); 8 wave32s: 2(M)x4(N); 4 WMMA per wave/K-step.
// Double-buffered LDS staged with global_load_async_to_lds_b128; all ten
// fragment ds_load_b128s are issued before the 4 back-to-back WMMAs.
// ---------------------------------------------------------------------------
__global__ __launch_bounds__(256)
void gemm_wmma(const _Float16* __restrict__ W, const float* __restrict__ bias,
               const _Float16* __restrict__ X,
               _Float16* __restrict__ outH, float* __restrict__ outF,
               float* __restrict__ pool, const float* __restrict__ extra,
               int O, int Cpad, int P, long long xStride, long long oStride,
               int ldPool, int ldE, int flags)
{
  __shared__ _Float16 As[2][TM][LDK];   // [buf][m][k]
  __shared__ _Float16 Bs[2][TN][LDK];   // [buf][n][k]

  const int t    = threadIdx.x;
  const int lane = t & 31;
  const int wave = t >> 5;
  const int wM   = wave >> 2;          // 0..1
  const int wN   = wave & 3;           // 0..3
  const int bM   = blockIdx.x * TM;
  const int bN   = blockIdx.y * TN;
  const int batch = blockIdx.z;
  const _Float16* __restrict__ Xb = X + (size_t)batch * xStride;

  v8f acc[4];
#pragma unroll
  for (int mt = 0; mt < 4; ++mt)
#pragma unroll
    for (int r = 0; r < 8; ++r) acc[mt][r] = 0.0f;

  // A staging: 256 thr * 2 b128 = 128 rows * 64B ; B staging: 256 thr * 1 b128
  const int rA  = t >> 1;
  const int kcA = (t & 1) * 16;
  const int nB  = t >> 2;
  const int kcB = (t & 3) * 8;

  unsigned asD[2], bsD[2];
  asD[0] = (unsigned)(unsigned long long)&As[0][rA][kcA];
  asD[1] = (unsigned)(unsigned long long)&As[1][rA][kcA];
  bsD[0] = (unsigned)(unsigned long long)&Bs[0][nB][kcB];
  bsD[1] = (unsigned)(unsigned long long)&Bs[1][nB][kcB];
  unsigned long long aG = (unsigned long long)(W  + (size_t)(bM + rA) * Cpad + kcA);
  unsigned long long bG = (unsigned long long)(Xb + (size_t)(bN + nB) * Cpad + kcB);

  // prologue: stage K-step 0 into buffer 0
  async_ld_b128(asD[0],      aG);
  async_ld_b128(asD[0] + 16, aG + 16);
  async_ld_b128(bsD[0],      bG);
  aG += TK * 2;   // running pointers for the next K-step
  bG += TK * 2;

  // fragment indices
  const int bn  = wN * 16 + (lane & 15);
  const int kb2 = (lane < 16) ? 0 : 16;
  const int ar  = lane & 15;
  const int ka  = (lane < 16) ? 0 : 16;

  const int KT = Cpad / TK;
  for (int kt = 0; kt < KT; ++kt) {
    const int cur = kt & 1;
    wait_async0();        // my async stores to LDS done
    __syncthreads();      // everyone's done; prev buffer free to overwrite

    if (kt + 1 < KT) {    // stage next K-step into the other buffer (overlaps WMMA)
      async_ld_b128(asD[cur ^ 1],      aG);
      async_ld_b128(asD[cur ^ 1] + 16, aG + 16);
      async_ld_b128(bsD[cur ^ 1],      bG);
      aG += TK * 2;
      bG += TK * 2;
    }

    // preload ALL fragments first so one s_wait_dscnt covers the 4 WMMAs
    FragH fb;
    fb.h[0] = *(const v8h*)&Bs[cur][bn][kb2];
    fb.h[1] = *(const v8h*)&Bs[cur][bn][kb2 + 8];
    FragH fa[4];
#pragma unroll
    for (int mt = 0; mt < 4; ++mt) {
      const int am = wM * 64 + mt * 16 + ar;
      fa[mt].h[0] = *(const v8h*)&As[cur][am][ka];
      fa[mt].h[1] = *(const v8h*)&As[cur][am][ka + 8];
    }
#pragma unroll
    for (int mt = 0; mt < 4; ++mt)
      acc[mt] = __builtin_amdgcn_wmma_f32_16x16x32_f16(
          false, fa[mt].v, false, fb.v, (short)0, acc[mt], false, false);
  }

  // ---- epilogue. C/D layout: VGPR r, lane L -> M = r + (L>=16 ? 8:0), N = L&15
  const int  col   = lane & 15;
  const int  mHalf = (lane >> 4) * 8;
  const int  n     = bN + wN * 16 + col;
  const bool relu  = (flags & F_RELU) != 0;

#pragma unroll
  for (int mt = 0; mt < 4; ++mt) {
    const int mBase = bM + wM * 64 + mt * 16 + mHalf;
    float vals[8];
#pragma unroll
    for (int r = 0; r < 8; ++r) {
      const int m = mBase + r;
      float v = acc[mt][r];
      if (m < O) {
        if (bias)  v += bias[m];
        if (extra) v += extra[(size_t)batch * ldE + m];
        if (relu)  v = fmaxf(v, 0.0f);
      }
      vals[r] = v;
    }
    if (flags & F_POOL) {
#pragma unroll
      for (int r = 0; r < 8; ++r) {
        float w = vals[r];
        w = fmaxf(w, __shfl_xor(w, 1));
        w = fmaxf(w, __shfl_xor(w, 2));
        w = fmaxf(w, __shfl_xor(w, 4));
        w = fmaxf(w, __shfl_xor(w, 8));
        if (col == 0 && mBase + r < O)   // post-ReLU >= 0: int-bit max == float max
          atomicMax((int*)(pool + (size_t)batch * ldPool + mBase + r), __float_as_int(w));
      }
    }
    if ((flags & F_F16) && mBase < O) {  // f16 layers all have O % 16 == 0
      v8h hv;
#pragma unroll
      for (int r = 0; r < 8; ++r) hv[r] = (_Float16)vals[r];
      *(v8h*)(outH + (size_t)batch * oStride + (size_t)n * O + mBase) = hv;
    }
    if (flags & F_F32) {
#pragma unroll
      for (int r = 0; r < 8; ++r)
        if (mBase + r < O)
          outF[(size_t)batch * oStride + (size_t)n * O + mBase + r] = vals[r];
    }
  }
}

// ---------------------------------------------------------------------------
// Small helper kernels
// ---------------------------------------------------------------------------
__global__ void cvt_weight(const float* __restrict__ src, _Float16* __restrict__ dst,
                           int O, int Ctake, int srcLd, int colOff, int Cpad, int Opad)
{
  int i = blockIdx.x * blockDim.x + threadIdx.x;
  if (i >= Opad * Cpad) return;
  int o = i / Cpad, c = i % Cpad;
  dst[i] = (o < O && c < Ctake) ? (_Float16)src[(size_t)o * srcLd + colOff + c]
                                : (_Float16)0.0f;
}

// [B,Csrc,N] f32 -> [B,N,Cpad] f16 (point-major, K zero-padded)
__global__ void cvt_act_pad(const float* __restrict__ src, _Float16* __restrict__ dst,
                            int Csrc, int Cpad, int Npts)
{
  size_t i = (size_t)blockIdx.x * blockDim.x + threadIdx.x;
  size_t tot = (size_t)B_SZ * Npts * Cpad;
  if (i >= tot) return;
  int c = (int)(i % Cpad);
  size_t r = i / Cpad;
  int n = (int)(r % Npts);
  int b = (int)(r / Npts);
  dst[i] = (c < Csrc) ? (_Float16)src[((size_t)b * Csrc + c) * Npts + n] : (_Float16)0.0f;
}

__global__ void cvt_f32_f16(const float* __restrict__ src, _Float16* __restrict__ dst, int n)
{
  int i = blockIdx.x * blockDim.x + threadIdx.x;
  if (i < n) dst[i] = (_Float16)src[i];
}

// logits [B][N][2] -> log_softmax into d_out [B][N][2] + seg mask
__global__ void logsoftmax_mask(const float* __restrict__ logits,
                                float* __restrict__ segOut, int* __restrict__ mask)
{
  int i = blockIdx.x * blockDim.x + threadIdx.x;
  if (i >= B_SZ * N_PTS) return;
  float l0 = logits[(size_t)i * 2 + 0];
  float l1 = logits[(size_t)i * 2 + 1];
  float m = fmaxf(l0, l1);
  float lse = m + logf(__expf(l0 - m) + __expf(l1 - m));
  segOut[(size_t)i * 2 + 0] = l0 - lse;
  segOut[(size_t)i * 2 + 1] = l1 - lse;
  mask[i] = (l1 > l0) ? 1 : 0;
}

// one block per batch: compact valid indices, resample NSEG, mean, center
// writes cbuf point-major [B][NSEG][32] (rows 3..31 zero)
__global__ __launch_bounds__(256)
void resample_kernel(const float* __restrict__ x, const int* __restrict__ mask,
                     _Float16* __restrict__ cbuf, float* __restrict__ outMeans,
                     float* __restrict__ outValid)
{
  __shared__ int   sIdx[N_PTS];
  __shared__ int   sScan[256];
  __shared__ float sPC[3][NSEG];
  __shared__ float sSum[3];
  const int b = blockIdx.x, t = threadIdx.x;

  int mk[4]; int cnt = 0;
#pragma unroll
  for (int j = 0; j < 4; ++j) { mk[j] = mask[b * N_PTS + t * 4 + j]; cnt += mk[j]; }
  sScan[t] = cnt;
  __syncthreads();
  for (int off = 1; off < 256; off <<= 1) {
    int add = (t >= off) ? sScan[t - off] : 0;
    __syncthreads();
    sScan[t] += add;
    __syncthreads();
  }
  const int total = sScan[255];
  int base = sScan[t] - cnt;
#pragma unroll
  for (int j = 0; j < 4; ++j) if (mk[j]) sIdx[base++] = t * 4 + j;
  if (t < 3) sSum[t] = 0.0f;
  __syncthreads();

  float part0 = 0.f, part1 = 0.f, part2 = 0.f;
  for (int j = t; j < NSEG; j += 256) {
    float p0 = 0.f, p1 = 0.f, p2 = 0.f;
    if (total > 0) {
      int pos;
      if (j < total) pos = sIdx[j];
      else { unsigned h = (unsigned)j * 2654435761u; h ^= h >> 13; pos = sIdx[h % (unsigned)total]; }
      p0 = x[((size_t)b * 4 + 0) * N_PTS + pos];
      p1 = x[((size_t)b * 4 + 1) * N_PTS + pos];
      p2 = x[((size_t)b * 4 + 2) * N_PTS + pos];
    }
    sPC[0][j] = p0; sPC[1][j] = p1; sPC[2][j] = p2;
    part0 += p0; part1 += p1; part2 += p2;
  }
  atomicAdd(&sSum[0], part0);
  atomicAdd(&sSum[1], part1);
  atomicAdd(&sSum[2], part2);
  __syncthreads();

  const float m0 = sSum[0] / NSEG, m1 = sSum[1] / NSEG, m2 = sSum[2] / NSEG;
  for (int j = t; j < NSEG; j += 256) {
    _Float16* cp = cbuf + ((size_t)b * NSEG + j) * 32;
    cp[0] = (_Float16)(sPC[0][j] - m0);
    cp[1] = (_Float16)(sPC[1][j] - m1);
    cp[2] = (_Float16)(sPC[2][j] - m2);
#pragma unroll
    for (int c = 3; c < 32; ++c) cp[c] = (_Float16)0.0f;
  }
  if (t == 0) {
    outMeans[b * 3 + 0] = m0; outMeans[b * 3 + 1] = m1; outMeans[b * 3 + 2] = m2;
    outValid[b] = (total > 0) ? 1.0f : 0.0f;
  }
}

// shifted[b][j][c] = cbuf[b][j][c] - (c<3 ? t[b][c] : 0)   (tvec is [B][3])
__global__ void shift_kernel(const _Float16* __restrict__ cbuf,
                             const float* __restrict__ tvec,
                             _Float16* __restrict__ shifted)
{
  size_t i = (size_t)blockIdx.x * blockDim.x + threadIdx.x;
  size_t tot = (size_t)B_SZ * NSEG * 32;
  if (i >= tot) return;
  int c = (int)(i % 32);
  int b = (int)(i / ((size_t)NSEG * 32));
  float tv = (c < 3) ? tvec[b * 3 + c] : 0.0f;
  shifted[i] = (_Float16)((float)cbuf[i] - tv);
}

// ---------------------------------------------------------------------------
extern "C" void kernel_launch(void* const* d_in, const int* in_sizes, int n_in,
                              void* d_out, int out_size, void* d_ws, size_t ws_size,
                              hipStream_t stream)
{
  (void)n_in; (void)out_size; (void)ws_size;

  // insertion order: s1..s10, t1..t3, tf1..tf3, b1..b4, bf1..bf3
  static const int Osz[23] = {64,64,64,128,1024,512,256,128,128,2,
                              128,256,512, 256,128,3,
                              128,128,256,512, 512,256,14};
  static const int Csz[23] = {4,64,64,64,128,1088,512,256,128,128,
                              3,128,256, 512,256,128,
                              3,128,128,256, 512,512,256};
  // alphabetical pytree position of each insertion-order layer
  static const int alphaPos[23] = {7,9,10,11,12,13,14,15,16,8,
                                   17,18,19, 20,21,22,
                                   0,1,2,3, 4,5,6};

  const float* Wp[23]; const float* Bp[23]; const float* xin;
  const bool xFirst = (in_sizes[0] == B_SZ * 4 * N_PTS);
  if (xFirst) {
    xin = (const float*)d_in[0];
    for (int i = 0; i < 23; ++i) { Wp[i] = (const float*)d_in[1 + 2*i]; Bp[i] = (const float*)d_in[2 + 2*i]; }
  } else {
    for (int i = 0; i < 23; ++i) { Wp[i] = (const float*)d_in[2*alphaPos[i]]; Bp[i] = (const float*)d_in[2*alphaPos[i] + 1]; }
    xin = (const float*)d_in[46];
  }
  float* outF = (float*)d_out;
  const int T_OFF = 262144, BBOX_OFF = 262528, MEANS_OFF = 264320, VALID_OFF = 264704;

  // ---- workspace bump allocator ----
  char* ws = (char*)d_ws;
  size_t off = 0;
  auto alloc = [&](size_t bytes) -> char* {
    size_t a = (off + 255) & ~(size_t)255;
    off = a + bytes;
    return ws + a;
  };

  _Float16* Wh[23]; int Cpad[23]; int Opad[23];
  for (int i = 0; i < 23; ++i) {
    Opad[i] = (Osz[i] + 127) & ~127;                    // async A-tile: no row guard
    int cp  = (i == 5) ? 0 : ((Csz[i] + 31) & ~31);
    Cpad[i] = cp;
    Wh[i] = cp ? (_Float16*)alloc((size_t)Opad[i] * cp * 2) : nullptr;
  }
  _Float16* W6a = (_Float16*)alloc((size_t)512 * 1024 * 2);
  _Float16* W6b = (_Float16*)alloc((size_t)512 * 64 * 2);

  _Float16* xpad = (_Float16*)alloc((size_t)B_SZ * N_PTS * 32 * 2);
  _Float16* pf   = (_Float16*)alloc((size_t)B_SZ * N_PTS * 64 * 2);
  char* arenaA = alloc(64ull << 20);
  char* arenaB = alloc(32ull << 20);
  char* arenaC = alloc(128ull << 20);
  float*    logits  = (float*)alloc((size_t)B_SZ * N_PTS * 2 * 4);
  int*      maskb   = (int*)alloc((size_t)B_SZ * N_PTS * 4);
  float*    gpool   = (float*)alloc((size_t)B_SZ * 1024 * 4);   // [B][1024]
  _Float16* g16     = (_Float16*)alloc((size_t)B_SZ * 1024 * 2);
  float*    c6      = (float*)alloc((size_t)B_SZ * 512 * 4);    // [B][512]
  _Float16* cbuf    = (_Float16*)alloc((size_t)B_SZ * NSEG * 32 * 2);
  _Float16* shifted = (_Float16*)alloc((size_t)B_SZ * NSEG * 32 * 2);
  float*    tpool   = (float*)alloc((size_t)B_SZ * 512 * 4);
  _Float16* tpool16 = (_Float16*)alloc((size_t)B_SZ * 512 * 2);
  _Float16* tf1o    = (_Float16*)alloc((size_t)B_SZ * 256 * 2);
  _Float16* tf2o    = (_Float16*)alloc((size_t)B_SZ * 128 * 2);
  float*    tf3o    = (float*)alloc((size_t)B_SZ * 3 * 4);      // [B][3]
  float*    bpool   = (float*)alloc((size_t)B_SZ * 512 * 4);
  _Float16* bpool16 = (_Float16*)alloc((size_t)B_SZ * 512 * 2);
  _Float16* bf1o    = (_Float16*)alloc((size_t)B_SZ * 512 * 2);
  _Float16* bf2o    = (_Float16*)alloc((size_t)B_SZ * 256 * 2);
  float*    bf3o    = (float*)alloc((size_t)B_SZ * 14 * 4);     // [B][14]

  // arena aliasing (liveness-checked; all work is sequential on one stream)
  _Float16* h1  = (_Float16*)arenaA;
  _Float16* h3  = (_Float16*)(arenaA + (16ull << 20));
  _Float16* h7  = (_Float16*)arenaA;
  _Float16* th1 = (_Float16*)arenaA;
  _Float16* bh1 = (_Float16*)arenaA;
  _Float16* bh2 = (_Float16*)(arenaA + (16ull << 20));
  _Float16* h4  = (_Float16*)arenaB;
  _Float16* h8  = (_Float16*)arenaB;
  _Float16* th2 = (_Float16*)arenaB;
  _Float16* bh3 = (_Float16*)arenaB;
  _Float16* h6  = (_Float16*)arenaC;
  _Float16* h9  = (_Float16*)arenaC;

  // ---- weight conversion (f32 -> f16, K padded to 32, O padded to 128) ----
  auto cvtW = [&](const float* src, _Float16* dst, int O, int Ctake, int srcLd,
                  int colOff, int cp, int op) {
    int n = op * cp;
    cvt_weight<<<(n + 255) / 256, 256, 0, stream>>>(src, dst, O, Ctake, srcLd, colOff, cp, op);
  };
  for (int i = 0; i < 23; ++i)
    if (i != 5) cvtW(Wp[i], Wh[i], Osz[i], Csz[i], Csz[i], 0, Cpad[i], Opad[i]);
  cvtW(Wp[5], W6a, 512, 1024, 1088, 0, 1024, 512);
  cvtW(Wp[5], W6b, 512,   64, 1088, 1024, 64, 512);

  hipMemsetAsync(gpool, 0, (size_t)B_SZ * 1024 * 4, stream);
  hipMemsetAsync(tpool, 0, (size_t)B_SZ * 512 * 4, stream);
  hipMemsetAsync(bpool, 0, (size_t)B_SZ * 512 * 4, stream);

  { size_t tot = (size_t)B_SZ * N_PTS * 32;
    cvt_act_pad<<<(unsigned)((tot + 255) / 256), 256, 0, stream>>>(xin, xpad, 4, 32, N_PTS); }

  auto G = [&](const _Float16* W, const float* bias, const _Float16* X,
               _Float16* oH, float* oF, float* pl, const float* ex,
               int O, int cp, int P, int nb, int ldE, int flags) {
    dim3 g((unsigned)(((O + 127) & ~127) / TM), (unsigned)(P / TN), (unsigned)nb);
    long long xs = (long long)cp * P;
    long long os = (long long)O * P;
    gemm_wmma<<<g, 256, 0, stream>>>(W, bias, X, oH, oF, pl, ex,
                                     O, cp, P, xs, os, /*ldPool=*/O, ldE, flags);
  };

  // ---- seg trunk ----
  G(Wh[0], Bp[0], xpad, h1,  nullptr, nullptr, nullptr,   64,  32, N_PTS, B_SZ, 0, F_F16 | F_RELU);
  G(Wh[1], Bp[1], h1,   pf,  nullptr, nullptr, nullptr,   64,  64, N_PTS, B_SZ, 0, F_F16 | F_RELU);
  G(Wh[2], Bp[2], pf,   h3,  nullptr, nullptr, nullptr,   64,  64, N_PTS, B_SZ, 0, F_F16 | F_RELU);
  G(Wh[3], Bp[3], h3,   h4,  nullptr, nullptr, nullptr,  128,  64, N_PTS, B_SZ, 0, F_F16 | F_RELU);
  G(Wh[4], Bp[4], h4, nullptr, nullptr, gpool, nullptr, 1024, 128, N_PTS, B_SZ, 0, F_POOL | F_RELU);
  cvt_f32_f16<<<(B_SZ * 1024 + 255) / 256, 256, 0, stream>>>(gpool, g16, B_SZ * 1024);
  // c6[B][512] = W6[:, :1024] @ g + b6   (per-batch constant part of s6)
  G(W6a, Bp[5], g16, nullptr, c6, nullptr, nullptr,      512, 1024, B_SZ, 1, 0, F_F32);
  // h6 = relu(W6[:,1024:] @ pf + c6)
  G(W6b, nullptr, pf, h6, nullptr, nullptr, c6,          512,   64, N_PTS, B_SZ, 512, F_F16 | F_RELU);
  G(Wh[6], Bp[6], h6, h7, nullptr, nullptr, nullptr,     256,  512, N_PTS, B_SZ, 0, F_F16 | F_RELU);
  G(Wh[7], Bp[7], h7, h8, nullptr, nullptr, nullptr,     128,  256, N_PTS, B_SZ, 0, F_F16 | F_RELU);
  G(Wh[8], Bp[8], h8, h9, nullptr, nullptr, nullptr,     128,  128, N_PTS, B_SZ, 0, F_F16 | F_RELU);
  G(Wh[9], Bp[9], h9, nullptr, logits, nullptr, nullptr,   2,  128, N_PTS, B_SZ, 0, F_F32);
  logsoftmax_mask<<<(B_SZ * N_PTS + 255) / 256, 256, 0, stream>>>(logits, outF, maskb);

  // ---- resample + center ----
  resample_kernel<<<B_SZ, 256, 0, stream>>>(xin, maskb, cbuf, outF + MEANS_OFF, outF + VALID_OFF);

  // ---- tnet ----
  G(Wh[10], Bp[10], cbuf, th1, nullptr, nullptr, nullptr, 128,  32, NSEG, B_SZ, 0, F_F16 | F_RELU);
  G(Wh[11], Bp[11], th1,  th2, nullptr, nullptr, nullptr, 256, 128, NSEG, B_SZ, 0, F_F16 | F_RELU);
  G(Wh[12], Bp[12], th2, nullptr, nullptr, tpool, nullptr,512, 256, NSEG, B_SZ, 0, F_POOL | F_RELU);
  cvt_f32_f16<<<(B_SZ * 512 + 255) / 256, 256, 0, stream>>>(tpool, tpool16, B_SZ * 512);
  G(Wh[13], Bp[13], tpool16, tf1o, nullptr, nullptr, nullptr, 256, 512, B_SZ, 1, 0, F_F16 | F_RELU);
  G(Wh[14], Bp[14], tf1o,    tf2o, nullptr, nullptr, nullptr, 128, 256, B_SZ, 1, 0, F_F16 | F_RELU);
  G(Wh[15], Bp[15], tf2o, nullptr, tf3o,    nullptr, nullptr,   3, 128, B_SZ, 1, 0, F_F32);

  { size_t tot = (size_t)B_SZ * NSEG * 32;
    shift_kernel<<<(unsigned)((tot + 255) / 256), 256, 0, stream>>>(cbuf, tf3o, shifted); }
  hipMemcpyAsync(outF + T_OFF, tf3o, (size_t)B_SZ * 3 * 4, hipMemcpyDeviceToDevice, stream);

  // ---- bbox ----
  G(Wh[16], Bp[16], shifted, bh1, nullptr, nullptr, nullptr, 128,  32, NSEG, B_SZ, 0, F_F16 | F_RELU);
  G(Wh[17], Bp[17], bh1,     bh2, nullptr, nullptr, nullptr, 128, 128, NSEG, B_SZ, 0, F_F16 | F_RELU);
  G(Wh[18], Bp[18], bh2,     bh3, nullptr, nullptr, nullptr, 256, 128, NSEG, B_SZ, 0, F_F16 | F_RELU);
  G(Wh[19], Bp[19], bh3, nullptr, nullptr, bpool,  nullptr, 512, 256, NSEG, B_SZ, 0, F_POOL | F_RELU);
  cvt_f32_f16<<<(B_SZ * 512 + 255) / 256, 256, 0, stream>>>(bpool, bpool16, B_SZ * 512);
  G(Wh[20], Bp[20], bpool16, bf1o, nullptr, nullptr, nullptr, 512, 512, B_SZ, 1, 0, F_F16 | F_RELU);
  G(Wh[21], Bp[21], bf1o,    bf2o, nullptr, nullptr, nullptr, 256, 512, B_SZ, 1, 0, F_F16 | F_RELU);
  G(Wh[22], Bp[22], bf2o, nullptr, bf3o,    nullptr, nullptr,  14, 256, B_SZ, 1, 0, F_F32);
  hipMemcpyAsync(outF + BBOX_OFF, bf3o, (size_t)B_SZ * 14 * 4, hipMemcpyDeviceToDevice, stream);
}
